// MultiFrameTransformerBlock_17755394801894
// MI455X (gfx1250) — matrix operand order repaired and checked by reading, not compiled, not run on hardware
//
#include <hip/hip_runtime.h>
#include <hip/hip_bf16.h>

#define BTc 8
#define Nc  2048
#define DMc 128
#define DPc 64
#define KSc 16
#define KTc 8
#define SCALEc 0.08838834764831845f  // 1/sqrt(128)

typedef __attribute__((ext_vector_type(16))) __bf16 v16bf;
typedef __attribute__((ext_vector_type(8)))  float  v8f;

__device__ __forceinline__ unsigned short f2bf(float f) {
  unsigned int u = __float_as_uint(f);
  unsigned int r = (u + 0x7FFFu + ((u >> 16) & 1u)) >> 16;
  return (unsigned short)r;
}
__device__ __forceinline__ unsigned pack2bf(float a, float b) {
  return (unsigned)f2bf(a) | ((unsigned)f2bf(b) << 16);
}

// A-frag: 16x32 bf16 tile, rows at `base` (row stride ld elements, ld multiple of 8), K offset k0 (mult 32).
// wave32 layout: lanes 0-15 row=lane K:{0..7,16..23}; lanes 16-31 row=lane-16 K:{8..15,24..31}
__device__ __forceinline__ v16bf load_a_bf16(const unsigned short* base, int ld, int lane, int k0) {
  int half = (lane >> 4) & 1;
  int row  = lane & 15;
  const unsigned short* rp = base + row * ld + k0 + (half ? 8 : 0);
  union { v16bf v; uint4 q[2]; } f;
  f.q[0] = *(const uint4*)(rp);
  f.q[1] = *(const uint4*)(rp + 16);
  return f.v;
}

// B-frag: 32x16 bf16 tile where B[k][n] = W[n][k] (W row-major, stride ld mult of 8).
// lanes 0-15: col n=lane, K:{0..7,16..23} ; lanes 16-31: col n=lane-16, K:{8..15,24..31}
__device__ __forceinline__ v16bf load_b_bf16(const unsigned short* w, int ld, int lane, int col0, int k0) {
  int half = (lane >> 4) & 1;
  int col  = lane & 15;
  const unsigned short* rp = w + (long)(col0 + col) * ld + k0 + (half ? 8 : 0);
  union { v16bf v; uint4 q[2]; } f;
  f.q[0] = *(const uint4*)(rp);
  f.q[1] = *(const uint4*)(rp + 16);
  return f.v;
}

__device__ __forceinline__ v8f wmma_bf16(v16bf a, v16bf b, v8f c) {
  return __builtin_amdgcn_wmma_f32_16x16x32_bf16(false, a, false, b, (short)0, c, false, false);
}

// Async global->LDS copy of 32B per lane (2 x b128), tracked by ASYNCcnt.
__device__ __forceinline__ void async_stage32(const void* g, unsigned lds) {
  unsigned long long ga = (unsigned long long)g;
  asm volatile("global_load_async_to_lds_b128 %0, %1, off" :: "v"(lds), "v"(ga) : "memory");
  asm volatile("global_load_async_to_lds_b128 %0, %1, off offset:16" :: "v"(lds), "v"(ga) : "memory");
}

// ---------------- conversion ----------------
__global__ void cvt_bf16_kernel(const float* __restrict__ src, unsigned short* __restrict__ dst, int n) {
  int i = blockIdx.x * blockDim.x + threadIdx.x;
  if (i < n) dst[i] = f2bf(src[i]);
}

// ---------------- fused x = f@W1^T+b1 ; q/k/v = x@W^T ----------------
__global__ void qkv_kernel(const unsigned short* __restrict__ fbf,
                           const unsigned short* __restrict__ W1bf, const float* __restrict__ b1,
                           const unsigned short* __restrict__ Wqbf, const unsigned short* __restrict__ Wkbf,
                           const unsigned short* __restrict__ Wvbf,
                           float* __restrict__ qg, float* __restrict__ kg, float* __restrict__ vg) {
  __shared__ alignas(16) unsigned short xb[16 * DMc];
  int lane = threadIdx.x & 31;
  int tile = blockIdx.x;              // BTc * 128
  int bt   = tile >> 7;
  int n0   = (tile & 127) << 4;
  int half = lane >> 4, col = lane & 15;

  const unsigned short* arow = fbf + (long)(bt * Nc + n0) * DPc;
  v16bf a0 = load_a_bf16(arow, DPc, lane, 0);
  v16bf a1 = load_a_bf16(arow, DPc, lane, 32);

#pragma unroll
  for (int ct = 0; ct < 8; ++ct) {
    v8f c = {};
    c = wmma_bf16(a0, load_b_bf16(W1bf, DPc, lane, ct * 16, 0), c);
    c = wmma_bf16(a1, load_b_bf16(W1bf, DPc, lane, ct * 16, 32), c);
    float bias = b1[ct * 16 + col];
#pragma unroll
    for (int r = 0; r < 8; ++r) {
      int m = r + (half ? 8 : 0);
      xb[m * DMc + ct * 16 + col] = f2bf(c[r] + bias);
    }
  }
  __syncthreads();

  v16bf xa[4];
#pragma unroll
  for (int i = 0; i < 4; ++i) xa[i] = load_a_bf16(xb, DMc, lane, 32 * i);

  const unsigned short* Ws[3] = {Wqbf, Wkbf, Wvbf};
  float* outs[3] = {qg, kg, vg};
#pragma unroll
  for (int w = 0; w < 3; ++w) {
    float* og = outs[w] + (long)(bt * Nc + n0) * DMc;
#pragma unroll
    for (int ct = 0; ct < 8; ++ct) {
      v8f c = {};
#pragma unroll
      for (int kk = 0; kk < 4; ++kk)
        c = wmma_bf16(xa[kk], load_b_bf16(Ws[w], DMc, lane, ct * 16, 32 * kk), c);
#pragma unroll
      for (int r = 0; r < 8; ++r) {
        int m = r + (half ? 8 : 0);
        og[m * DMc + ct * 16 + col] = c[r];
      }
    }
  }
}

// ---------------- spatial KNN: top-16 nearest (squared dist) ----------------
__global__ void knn_kernel(const float* __restrict__ xyz, int* __restrict__ sidx) {
  __shared__ float sx[256 * 3];
  int t  = threadIdx.x;
  int bt = blockIdx.x >> 3;
  int n  = ((blockIdx.x & 7) << 8) + t;
  const float* fx = xyz + (long)bt * Nc * 3;
  float qx = fx[n * 3 + 0], qy = fx[n * 3 + 1], qz = fx[n * 3 + 2];
  float bd[KSc]; int bi[KSc];
#pragma unroll
  for (int i = 0; i < KSc; ++i) { bd[i] = 3.4e38f; bi[i] = 0; }
  for (int tile = 0; tile < Nc; tile += 256) {
    __syncthreads();
    for (int j = t; j < 256 * 3; j += 256) sx[j] = fx[tile * 3 + j];
    __syncthreads();
    for (int j = 0; j < 256; ++j) {
      float dx = qx - sx[j * 3 + 0], dy = qy - sx[j * 3 + 1], dz = qz - sx[j * 3 + 2];
      float d = dx * dx + dy * dy + dz * dz;
      if (d < bd[KSc - 1]) {
        int p = KSc - 1;
        while (p > 0 && bd[p - 1] > d) { bd[p] = bd[p - 1]; bi[p] = bi[p - 1]; --p; }
        bd[p] = d; bi[p] = tile + j;
      }
    }
  }
  for (int i = 0; i < KSc; ++i) sidx[(long)(bt * Nc + n) * KSc + i] = bi[i];
}

// ---------------- spatial attention (4 queries x 16 neighbors per block) ----------------
#define HP 136   // bf16 LDS row stride for h  (136*2 = 272 B, 16B multiple)
#define PP 132   // f32 LDS row stride for pos (132*4 = 528 B, 16B multiple)
__global__ void spatial_attn_kernel(const float* __restrict__ xyz, const int* __restrict__ sidx,
                                    const float* __restrict__ qg, const float* __restrict__ kfg,
                                    const float* __restrict__ vfg,
                                    const float* __restrict__ Wd1, const float* __restrict__ bd1,
                                    const unsigned short* __restrict__ Wd2bf, const float* __restrict__ bd2,
                                    float* __restrict__ sfg, unsigned short* __restrict__ nrmg) {
  __shared__ alignas(16) unsigned short hls[64 * HP];
  __shared__ alignas(16) float pls[64 * PP];
  __shared__ float scl[64];
  __shared__ float atl[64];
  __shared__ int   sid[64];
  __shared__ alignas(16) float sfl[4 * DMc];
  __shared__ float invl[4];

  int t    = threadIdx.x;        // 128 threads = 4 waves
  int lane = t & 31;
  int tile = blockIdx.x;         // BTc * 512
  int bt   = tile >> 9;
  int q0   = (tile & 511) << 2;  // 4 queries
  const float* fx = xyz + (long)bt * Nc * 3;

  if (t < 64) sid[t] = sidx[(long)(bt * Nc + q0) * KSc + t];
  __syncthreads();

  // h = relu(delta @ Wd1^T + bd1) -> LDS bf16 (packed pair stores)
  {
    int p = t >> 1, hh = t & 1;
    int qq = p >> 4;
    int idx = sid[p];
    float dx = fx[(q0 + qq) * 3 + 0] - fx[idx * 3 + 0];
    float dy = fx[(q0 + qq) * 3 + 1] - fx[idx * 3 + 1];
    float dz = fx[(q0 + qq) * 3 + 2] - fx[idx * 3 + 2];
    for (int d = hh * 64; d < hh * 64 + 64; d += 2) {
      float v0 = bd1[d]     + dx * Wd1[d * 3 + 0] + dy * Wd1[d * 3 + 1] + dz * Wd1[d * 3 + 2];
      float v1 = bd1[d + 1] + dx * Wd1[d * 3 + 3] + dy * Wd1[d * 3 + 4] + dz * Wd1[d * 3 + 5];
      v0 = v0 > 0.f ? v0 : 0.f;
      v1 = v1 > 0.f ? v1 : 0.f;
      *(unsigned*)(hls + p * HP + d) = pack2bf(v0, v1);
    }
  }
  __syncthreads();

  // pos = h @ Wd2^T + bd2 via WMMA; 4 waves, one 16-row tile each
  {
    int wv = t >> 5;
    const unsigned short* hb = hls + wv * 16 * HP;
    v16bf a[4];
#pragma unroll
    for (int i = 0; i < 4; ++i) a[i] = load_a_bf16(hb, HP, lane, 32 * i);
    int half = lane >> 4, col = lane & 15;
#pragma unroll
    for (int ct = 0; ct < 8; ++ct) {
      v8f c = {};
#pragma unroll
      for (int kk = 0; kk < 4; ++kk)
        c = wmma_bf16(a[kk], load_b_bf16(Wd2bf, DMc, lane, ct * 16, 32 * kk), c);
      float bias = bd2[ct * 16 + col];
#pragma unroll
      for (int r = 0; r < 8; ++r) {
        int m = wv * 16 + r + (half ? 8 : 0);
        pls[m * PP + ct * 16 + col] = c[r] + bias;
      }
    }
  }
  __syncthreads();

  // attention scores q . (k_g + pos), vectorized float4
  if (t < 64) {
    int qq = t >> 4;
    int idx = sid[t];
    const float4* qr = (const float4*)(qg + (long)(bt * Nc + q0 + qq) * DMc);
    const float4* kr = (const float4*)(kfg + (long)(bt * Nc + idx) * DMc);
    const float4* pr = (const float4*)(pls + t * PP);
    float s = 0.f;
    for (int d = 0; d < DMc / 4; ++d) {
      float4 qv = qr[d], kv = kr[d], pv = pr[d];
      s += qv.x * (kv.x + pv.x) + qv.y * (kv.y + pv.y) +
           qv.z * (kv.z + pv.z) + qv.w * (kv.w + pv.w);
    }
    scl[t] = s * SCALEc;
  }
  __syncthreads();
  if (t < 4) {
    float mx = -3.4e38f;
    for (int k = 0; k < KSc; ++k) mx = fmaxf(mx, scl[t * KSc + k]);
    float sum = 0.f;
    for (int k = 0; k < KSc; ++k) { float e = __expf(scl[t * KSc + k] - mx); atl[t * KSc + k] = e; sum += e; }
    float inv = 1.f / sum;
    for (int k = 0; k < KSc; ++k) atl[t * KSc + k] *= inv;
  }
  __syncthreads();

  // sf = attn . (v_g + pos)
  {
    int qq = t >> 5;
    int d0 = (t & 31) << 2;
    float4 acc = {0.f, 0.f, 0.f, 0.f};
    for (int k = 0; k < KSc; ++k) {
      int p = qq * KSc + k;
      float w = atl[p];
      float4 vv = *(const float4*)(vfg + (long)(bt * Nc + sid[p]) * DMc + d0);
      float4 pv = *(const float4*)(pls + p * PP + d0);
      acc.x += w * (vv.x + pv.x); acc.y += w * (vv.y + pv.y);
      acc.z += w * (vv.z + pv.z); acc.w += w * (vv.w + pv.w);
    }
    long ro = (long)(bt * Nc + q0 + qq) * DMc;
    *(float4*)(sfl + qq * DMc + d0) = acc;
    *(float4*)(sfg + ro + d0) = acc;
  }
  __syncthreads();
  if (t < 4) {
    float s2 = 0.f;
    for (int d = 0; d < DMc; ++d) { float v = sfl[t * DMc + d]; s2 += v * v; }
    invl[t] = 1.f / fmaxf(sqrtf(s2), 1e-12f);
  }
  __syncthreads();
  {
    int qq = t >> 5;
    int d0 = (t & 31) << 2;
    float inv = invl[qq];
    long ro = (long)(bt * Nc + q0 + qq) * DMc;
    unsigned* np = (unsigned*)(nrmg + ro + d0);
    np[0] = pack2bf(sfl[qq * DMc + d0 + 0] * inv, sfl[qq * DMc + d0 + 1] * inv);
    np[1] = pack2bf(sfl[qq * DMc + d0 + 2] * inv, sfl[qq * DMc + d0 + 3] * inv);
  }
}

// ---------------- cosine sim (WMMA, async-LDS double-buffered B) + top-8 ----------------
__global__ void sim_topk_kernel(const unsigned short* __restrict__ nrmg, int* __restrict__ tidx) {
  __shared__ alignas(16) unsigned short bstage[2][16 * DMc];  // 2 x 4 KB staged col-tiles
  __shared__ float tb[64 * 17];
  int t    = threadIdx.x;       // 128 threads = 4 waves
  int lane = t & 31;
  int wv   = t >> 5;
  int blk  = blockIdx.x;        // BTc * 32
  int bt   = blk >> 5;
  int r0   = (blk & 31) << 6;   // 64 rows per block (16 per wave)
  const unsigned short* nb = nrmg + (long)bt * Nc * DMc;

  v16bf a[4];
#pragma unroll
  for (int i = 0; i < 4; ++i)
    a[i] = load_a_bf16(nb + (long)(r0 + wv * 16) * DMc, DMc, lane, 32 * i);
  int half = lane >> 4, col = lane & 15;

  float bs[KTc]; int bix[KTc];
#pragma unroll
  for (int i = 0; i < KTc; ++i) { bs[i] = -3.4e38f; bix[i] = 0; }

  unsigned l0 = (unsigned)(unsigned long long)&bstage[0][0] + t * 32;
  unsigned l1 = (unsigned)(unsigned long long)&bstage[1][0] + t * 32;

  // prologue: stage col-tile 0 (16 rows x 128 bf16 = 4 KB, 32 B per thread)
  async_stage32((const char*)nb + t * 32, l0);

  for (int ct = 0; ct < Nc / 16; ++ct) {
    if (ct + 1 < Nc / 16) {
      const char* src = (const char*)(nb + (long)((ct + 1) * 16) * DMc) + t * 32;
      async_stage32(src, ((ct + 1) & 1) ? l1 : l0);
      asm volatile("s_wait_asynccnt 0x2" ::: "memory");  // tile ct landed; ct+1 in flight
    } else {
      asm volatile("s_wait_asynccnt 0x0" ::: "memory");
    }
    __syncthreads();

    const unsigned short* bb = (ct & 1) ? &bstage[1][0] : &bstage[0][0];
    v8f c = {};
#pragma unroll
    for (int kk = 0; kk < 4; ++kk)
      c = wmma_bf16(a[kk], load_b_bf16(bb, DMc, lane, 0, 32 * kk), c);
#pragma unroll
    for (int r = 0; r < 8; ++r)
      tb[(wv * 16 + r + (half ? 8 : 0)) * 17 + col] = c[r];
    __syncthreads();

    if (t < 64) {
      for (int j = 0; j < 16; ++j) {
        float v = tb[t * 17 + j];
        if (v > bs[KTc - 1]) {
          int p = KTc - 1;
          while (p > 0 && bs[p - 1] < v) { bs[p] = bs[p - 1]; bix[p] = bix[p - 1]; --p; }
          bs[p] = v; bix[p] = ct * 16 + j;
        }
      }
    }
    __syncthreads();
  }
  if (t < 64)
    for (int i = 0; i < KTc; ++i) tidx[(long)(bt * Nc + r0 + t) * KTc + i] = bix[i];
}

// ---------------- temporal attention + output projection (+transpose) ----------------
__global__ void temporal_out_kernel(const float* __restrict__ sfg, const int* __restrict__ tidx,
                                    const float* __restrict__ W2, const float* __restrict__ b2,
                                    float* __restrict__ outg) {
  __shared__ float sc[16 * KTc];
  __shared__ float at[16 * KTc];
  __shared__ alignas(16) float tfl[16 * DMc];
  int t = threadIdx.x;             // 256 threads = 16 queries x 16
  int g = t >> 4, j = t & 15;
  int blk = blockIdx.x;            // BTc * 128
  int bt  = blk >> 7;
  int n   = (blk & 127) * 16 + g;
  const float* sfb = sfg + (long)bt * Nc * DMc;
  const int* tix = tidx + (long)(bt * Nc + n) * KTc;

  if (j < KTc) {
    const float4* a4 = (const float4*)(sfb + (long)n * DMc);
    const float4* b4 = (const float4*)(sfb + (long)tix[j] * DMc);
    float s = 0.f;
    for (int d = 0; d < DMc / 4; ++d) {
      float4 av = a4[d], bv = b4[d];
      s += av.x * bv.x + av.y * bv.y + av.z * bv.z + av.w * bv.w;
    }
    sc[g * KTc + j] = s * SCALEc;
  }
  __syncthreads();
  if (j == 0) {
    float mx = -3.4e38f;
    for (int k = 0; k < KTc; ++k) mx = fmaxf(mx, sc[g * KTc + k]);
    float e[KTc]; float sum = 0.f;
    for (int k = 0; k < KTc; ++k) { e[k] = __expf(sc[g * KTc + k] - mx); sum += e[k]; }
    for (int k = 0; k < KTc; ++k) at[g * KTc + k] = e[k] / sum;
  }
  __syncthreads();
  {
    int d0 = j * 8;
    float4 acc0 = {0, 0, 0, 0}, acc1 = {0, 0, 0, 0};
    for (int k = 0; k < KTc; ++k) {
      float w = at[g * KTc + k];
      const float4* b4 = (const float4*)(sfb + (long)tix[k] * DMc + d0);
      float4 u = b4[0], v = b4[1];
      acc0.x += w * u.x; acc0.y += w * u.y; acc0.z += w * u.z; acc0.w += w * u.w;
      acc1.x += w * v.x; acc1.y += w * v.y; acc1.z += w * v.z; acc1.w += w * v.w;
    }
    *(float4*)(tfl + g * DMc + d0) = acc0;
    *(float4*)(tfl + g * DMc + d0 + 4) = acc1;
  }
  __syncthreads();
  {
    const float4* a4 = (const float4*)(sfb + (long)n * DMc);
    const float4* t4 = (const float4*)(tfl + g * DMc);
    for (int oo = 0; oo < 4; ++oo) {
      int od = j * 4 + oo;
      const float4* w4 = (const float4*)(W2 + (long)od * (2 * DMc));
      float acc = b2[od];
      for (int d = 0; d < DMc / 4; ++d) {
        float4 av = a4[d], wv = w4[d];
        acc += av.x * wv.x + av.y * wv.y + av.z * wv.z + av.w * wv.w;
      }
      for (int d = 0; d < DMc / 4; ++d) {
        float4 tv = t4[d], wv = w4[DMc / 4 + d];
        acc += tv.x * wv.x + tv.y * wv.y + tv.z * wv.z + tv.w * wv.w;
      }
      outg[((long)bt * DPc + od) * Nc + n] = acc;
    }
  }
}

extern "C" void kernel_launch(void* const* d_in, const int* in_sizes, int n_in,
                              void* d_out, int out_size, void* d_ws, size_t ws_size,
                              hipStream_t stream) {
  (void)in_sizes; (void)n_in; (void)out_size; (void)ws_size;
  const float* features = (const float*)d_in[0];
  const float* xyz = (const float*)d_in[1];
  const float* W1  = (const float*)d_in[2];
  const float* b1  = (const float*)d_in[3];
  const float* Wq  = (const float*)d_in[4];
  const float* Wk  = (const float*)d_in[5];
  const float* Wv  = (const float*)d_in[6];
  const float* Wd1 = (const float*)d_in[7];
  const float* bd1 = (const float*)d_in[8];
  const float* Wd2 = (const float*)d_in[9];
  const float* bd2 = (const float*)d_in[10];
  const float* W2  = (const float*)d_in[11];
  const float* b2  = (const float*)d_in[12];
  float* outg = (float*)d_out;

  char* ws = (char*)d_ws;
  size_t off = 0;
  auto carve = [&](size_t bytes) -> void* {
    void* p = ws + off;
    off += (bytes + 255) & ~(size_t)255;
    return p;
  };
  unsigned short* fbf   = (unsigned short*)carve((size_t)BTc * Nc * DPc * 2);
  unsigned short* W1bf  = (unsigned short*)carve((size_t)DMc * DPc * 2);
  unsigned short* Wqbf  = (unsigned short*)carve((size_t)DMc * DMc * 2);
  unsigned short* Wkbf  = (unsigned short*)carve((size_t)DMc * DMc * 2);
  unsigned short* Wvbf  = (unsigned short*)carve((size_t)DMc * DMc * 2);
  unsigned short* Wd2bf = (unsigned short*)carve((size_t)DMc * DMc * 2);
  float* qg  = (float*)carve((size_t)BTc * Nc * DMc * 4);
  float* kfg = (float*)carve((size_t)BTc * Nc * DMc * 4);
  float* vfg = (float*)carve((size_t)BTc * Nc * DMc * 4);
  float* sfg = (float*)carve((size_t)BTc * Nc * DMc * 4);
  unsigned short* nrmg = (unsigned short*)carve((size_t)BTc * Nc * DMc * 2);
  int* sidxg = (int*)carve((size_t)BTc * Nc * KSc * 4);
  int* tidxg = (int*)carve((size_t)BTc * Nc * KTc * 4);

  int nf = BTc * Nc * DPc;
  cvt_bf16_kernel<<<(nf + 255) / 256, 256, 0, stream>>>(features, fbf, nf);
  cvt_bf16_kernel<<<(DMc * DPc + 255) / 256, 256, 0, stream>>>(W1, W1bf, DMc * DPc);
  cvt_bf16_kernel<<<(DMc * DMc + 255) / 256, 256, 0, stream>>>(Wq, Wqbf, DMc * DMc);
  cvt_bf16_kernel<<<(DMc * DMc + 255) / 256, 256, 0, stream>>>(Wk, Wkbf, DMc * DMc);
  cvt_bf16_kernel<<<(DMc * DMc + 255) / 256, 256, 0, stream>>>(Wv, Wvbf, DMc * DMc);
  cvt_bf16_kernel<<<(DMc * DMc + 255) / 256, 256, 0, stream>>>(Wd2, Wd2bf, DMc * DMc);

  qkv_kernel<<<BTc * 128, 32, 0, stream>>>(fbf, W1bf, b1, Wqbf, Wkbf, Wvbf, qg, kfg, vfg);
  knn_kernel<<<BTc * 8, 256, 0, stream>>>(xyz, sidxg);
  spatial_attn_kernel<<<BTc * 512, 128, 0, stream>>>(xyz, sidxg, qg, kfg, vfg,
                                                     Wd1, bd1, Wd2bf, bd2, sfg, nrmg);
  sim_topk_kernel<<<BTc * 32, 128, 0, stream>>>(nrmg, tidxg);
  temporal_out_kernel<<<BTc * 128, 256, 0, stream>>>(sfg, tidxg, W2, b2, outg);
}
